// VectorCollapseEngine_79800492359838
// MI455X (gfx1250) — compile-verified, matrix operand order checked by reading, not run on hardware
//
#include <hip/hip_runtime.h>
#include <math.h>

#define DIMX 1024
#define BATCH 65536
#define NLAYERS 6

typedef __attribute__((ext_vector_type(16))) __bf16 v16bf;
typedef __attribute__((ext_vector_type(8)))  __bf16 v8bf;
typedef __attribute__((ext_vector_type(8)))  float  v8f;

__device__ __forceinline__ __bf16 tobf(float x) { return (__bf16)x; }

// Branch-free tanh: (e^{2x}-1)/(e^{2x}+1), e^{2x} = 2^(x*2*log2(e)).
__device__ __forceinline__ float fast_tanh(float x) {
  float xc = fminf(fmaxf(x, -9.f), 9.f);
  float e = __builtin_amdgcn_exp2f(xc * 2.88539008177793f);  // exp(2x)
  return (e - 1.f) * __builtin_amdgcn_rcpf(e + 1.f);
}

// ---------------------------------------------------------------------------
// Normalize the three anchor vectors -> dirs[3][DIMX]
// ---------------------------------------------------------------------------
__global__ __launch_bounds__(256) void prep_dirs(
    const float* __restrict__ ae, const float* __restrict__ ac,
    const float* __restrict__ an, float* __restrict__ dirs) {
  const float* src = (blockIdx.x == 0) ? ae : ((blockIdx.x == 1) ? ac : an);
  float* dst = dirs + (size_t)blockIdx.x * DIMX;
  __shared__ float red[256];
  float s = 0.f;
  for (int i = threadIdx.x; i < DIMX; i += 256) { float v = src[i]; s += v * v; }
  red[threadIdx.x] = s;
  __syncthreads();
  for (int off = 128; off > 0; off >>= 1) {
    if ((int)threadIdx.x < off) red[threadIdx.x] += red[threadIdx.x + off];
    __syncthreads();
  }
  float inv = 1.f / fmaxf(sqrtf(red[0]), 1e-12f);
  for (int i = threadIdx.x; i < DIMX; i += 256) dst[i] = src[i] * inv;
}

// ---------------------------------------------------------------------------
// Pack W (row-major [n][k], used as h@W^T) into bf16 WMMA B-fragment order:
// Wp[((nt*32 + kb)*32 + lane)*16 + g*8 + j]
//   n = nt*16 + (lane&15); half = lane>>4; k = kb*32 + g*16 + half*8 + j
// ---------------------------------------------------------------------------
__global__ __launch_bounds__(256) void pack_w(
    const float* __restrict__ W1, const float* __restrict__ W2,
    __bf16* __restrict__ W1p, __bf16* __restrict__ W2p) {
  const float* W = blockIdx.y ? W2 : W1;
  __bf16* Wp = blockIdx.y ? W2p : W1p;
  int idx = blockIdx.x * 256 + threadIdx.x;  // 0..65535
  int lane = idx & 31, kb = (idx >> 5) & 31, nt = idx >> 10;
  int n = nt * 16 + (lane & 15);
  int half = lane >> 4;
  __bf16* dst = Wp + ((size_t)(nt * 32 + kb) * 32 + lane) * 16;
#pragma unroll
  for (int g = 0; g < 2; ++g) {
    int k = kb * 32 + g * 16 + half * 8;
    const float* srow = W + (size_t)n * DIMX + k;
    v8bf t;
#pragma unroll
    for (int j = 0; j < 8; ++j) t[j] = tobf(srow[j]);
    *(v8bf*)(dst + g * 8) = t;
  }
}

// ---------------------------------------------------------------------------
// Per-row pass: one wave per row. Optional norm clamp (in place), trace
// outputs for this layer, and repulsion coefficients g_i for the epilogue.
// ---------------------------------------------------------------------------
__global__ __launch_bounds__(256) void rowpass(
    float* __restrict__ H, const float* __restrict__ dirs,
    float* __restrict__ Ge, float* __restrict__ Gc, float* __restrict__ Gn,
    float* __restrict__ out, int traceLayer, int doClamp) {
  int wave = threadIdx.x >> 5, lane = threadIdx.x & 31;
  int row = blockIdx.x * 8 + wave;
  float* p = H + (size_t)row * DIMX;
  const float* E = dirs;
  const float* C = dirs + DIMX;
  const float* N = dirs + 2 * DIMX;

  float4 r[8];
  float ss = 0.f, de = 0.f, dc = 0.f, dn = 0.f;
#pragma unroll
  for (int i = 0; i < 8; ++i) {
    int k4 = lane * 8 + i;
    r[i] = ((const float4*)p)[k4];
    float4 e = ((const float4*)E)[k4];
    float4 c = ((const float4*)C)[k4];
    float4 n = ((const float4*)N)[k4];
    ss += r[i].x * r[i].x + r[i].y * r[i].y + r[i].z * r[i].z + r[i].w * r[i].w;
    de += r[i].x * e.x + r[i].y * e.y + r[i].z * e.z + r[i].w * e.w;
    dc += r[i].x * c.x + r[i].y * c.y + r[i].z * c.z + r[i].w * c.w;
    dn += r[i].x * n.x + r[i].y * n.y + r[i].z * n.z + r[i].w * n.w;
  }
#pragma unroll
  for (int off = 16; off > 0; off >>= 1) {
    ss += __shfl_xor(ss, off, 32);
    de += __shfl_xor(de, off, 32);
    dc += __shfl_xor(dc, off, 32);
    dn += __shfl_xor(dn, off, 32);
  }
  float nrm = sqrtf(ss);
  float inn = 1.f / fmaxf(nrm, 1e-12f);
  float a_e = de * inn, a_c = dc * inn, a_n = dn * inn;
  float d_e = (1.f - a_e) * 0.5f, d_c = (1.f - a_c) * 0.5f,
        d_n = (1.f - a_n) * 0.5f;

  float scale = 1.f;
  if (doClamp && nrm > 10.f) scale = 10.f / (nrm + 1e-8f);
  if (doClamp) {
#pragma unroll
    for (int i = 0; i < 8; ++i) {
      r[i].x *= scale; r[i].y *= scale; r[i].z *= scale; r[i].w *= scale;
      ((float4*)p)[lane * 8 + i] = r[i];
    }
  }
  // coefficients use the CLAMPED h (a_i, d_i are scale-invariant)
  float ss2 = ss * scale * scale;
  float ie = 1.f / fmaxf(sqrtf(fmaxf(ss2 - 2.f * de * scale + 1.f, 0.f)), 1e-12f);
  float ic = 1.f / fmaxf(sqrtf(fmaxf(ss2 - 2.f * dc * scale + 1.f, 0.f)), 1e-12f);
  float in_ = 1.f / fmaxf(sqrtf(fmaxf(ss2 - 2.f * dn * scale + 1.f, 0.f)), 1e-12f);

  if (lane == 0) {
    Ge[row] = 0.10f * d_e * ie;
    Gc[row] = 0.10f * d_c * ic;
    Gn[row] = 0.05f * d_n * in_;
    if (traceLayer >= 0) {
      float* tb = out + (size_t)BATCH * DIMX;
      size_t o = (size_t)traceLayer * BATCH + row;
      size_t LB = (size_t)NLAYERS * BATCH;
      tb[o] = a_e;           tb[LB + o] = a_c;        tb[2 * LB + o] = a_n;
      tb[3 * LB + o] = d_e;  tb[4 * LB + o] = d_c;    tb[5 * LB + o] = d_n;
      tb[6 * LB + o] = d_e * d_e;
      tb[7 * LB + o] = d_c * d_c;
      tb[8 * LB + o] = d_n * d_n;
    }
  }
}

// ---------------------------------------------------------------------------
// GEMM1: T = tanh(h @ W1^T + b1).  128x128 tile / 256 threads (8 waves).
// Double-buffered LDS staging of A (f32->bf16), register prefetch of the next
// K-chunk; A-fragment LDS loads software-pipelined one subtile ahead of the
// WMMA burst (only 2 fragments live -> low pressure, non-zero dscnt waits).
// ---------------------------------------------------------------------------
__global__ __launch_bounds__(256) void gemm1_tanh(
    const float* __restrict__ A, const __bf16* __restrict__ Wp,
    const float* __restrict__ bias, __bf16* __restrict__ Tp) {
  __shared__ __align__(32) __bf16 sA[2][8 * 32 * 16];  // 2 x 8 KB
  __shared__ __align__(16) __bf16 sTile[8][256];       // 4 KB
  int tid = threadIdx.x;
  int wave = tid >> 5, lane = tid & 31, half = lane >> 4;
  int ntg = blockIdx.y * 8 + wave;

  v8f acc[8];
  v8f z = {0.f, 0.f, 0.f, 0.f, 0.f, 0.f, 0.f, 0.f};
#pragma unroll
  for (int i = 0; i < 8; ++i) acc[i] = z;

  // staging role: this thread fills A-fragment slot (mt = wave, lane)
  int mStage = blockIdx.x * 128 + wave * 16 + (lane & 15);
  const float* arow = A + (size_t)mStage * DIMX + half * 8;

  // prefetch K-chunk 0 into registers
  float4 f0a = *(const float4*)(arow);
  float4 f0b = *(const float4*)(arow + 4);
  float4 f1a = *(const float4*)(arow + 16);
  float4 f1b = *(const float4*)(arow + 20);

  for (int kb = 0; kb < 32; ++kb) {
    int buf = kb & 1;
    {
      v8bf t0, t1;
      t0[0] = tobf(f0a.x); t0[1] = tobf(f0a.y); t0[2] = tobf(f0a.z); t0[3] = tobf(f0a.w);
      t0[4] = tobf(f0b.x); t0[5] = tobf(f0b.y); t0[6] = tobf(f0b.z); t0[7] = tobf(f0b.w);
      t1[0] = tobf(f1a.x); t1[1] = tobf(f1a.y); t1[2] = tobf(f1a.z); t1[3] = tobf(f1a.w);
      t1[4] = tobf(f1b.x); t1[5] = tobf(f1b.y); t1[6] = tobf(f1b.z); t1[7] = tobf(f1b.w);
      __bf16* d = &sA[buf][(wave * 32 + lane) * 16];
      *(v8bf*)d = t0;
      *(v8bf*)(d + 8) = t1;
    }
    // prefetch next K-chunk while this one is consumed
    if (kb < 31) {
      const float* nx = arow + (kb + 1) * 32;
      f0a = *(const float4*)(nx);
      f0b = *(const float4*)(nx + 4);
      f1a = *(const float4*)(nx + 16);
      f1b = *(const float4*)(nx + 20);
    }
    __syncthreads();
    v16bf bfr = *(const v16bf*)(Wp + ((size_t)(ntg * 32 + kb) * 32 + lane) * 16);
    // software pipeline: fragment mt+1 loads in flight during WMMA mt
    v16bf a_cur = *(const v16bf*)(&sA[buf][(0 * 32 + lane) * 16]);
    v16bf a_nxt;
#pragma unroll
    for (int mt = 0; mt < 8; ++mt) {
      if (mt < 7)
        a_nxt = *(const v16bf*)(&sA[buf][((mt + 1) * 32 + lane) * 16]);
      acc[mt] = __builtin_amdgcn_wmma_f32_16x16x32_bf16(
          false, a_cur, false, bfr, (short)0, acc[mt], false, false);
      a_cur = a_nxt;
    }
  }

  // epilogue: + bias, tanh, shuffle C/D-layout tile into A-fragment layout
  int ncol = ntg * 16 + (lane & 15);
  float bv = bias[ncol];
  int kb2 = ntg >> 1, hsel = ntg & 1;  // T column range = K slice for GEMM2
#pragma unroll
  for (int mt = 0; mt < 8; ++mt) {
#pragma unroll
    for (int r = 0; r < 8; ++r) {
      float v = fast_tanh(acc[mt][r] + bv);
      // C/D layout: VGPR r, lanes 0-15 -> M=r, lanes 16-31 -> M=8+r
      sTile[wave][(r + half * 8) * 16 + (lane & 15)] = tobf(v);
    }
    // A layout read: m = lane&15, k_local = half*8 + 0..7 (same-wave DS order)
    v8bf tv = *(const v8bf*)(&sTile[wave][(lane & 15) * 16 + half * 8]);
    size_t mtg = (size_t)blockIdx.x * 8 + mt;
    *(v8bf*)(Tp + ((mtg * 32 + kb2) * 32 + lane) * 16 + hsel * 8) = tv;
  }
}

// ---------------------------------------------------------------------------
// GEMM2 + fused epilogue:
//   u = alpha*h + (T @ W2^T + b2) + ge*E[n] + gc*C[n] + gn*N[n],
//   alpha = 1 - (ge+gc+gn).  A read directly from packed T fragments with
//   one-subtile-ahead pipelining.
// ---------------------------------------------------------------------------
__global__ __launch_bounds__(256) void gemm2_epi(
    const __bf16* __restrict__ Tp, const __bf16* __restrict__ Wp,
    const float* __restrict__ bias, const float* __restrict__ Hin,
    const float* __restrict__ dirs, const float* __restrict__ Ge,
    const float* __restrict__ Gc, const float* __restrict__ Gn,
    float* __restrict__ Hout) {
  int tid = threadIdx.x;
  int wave = tid >> 5, lane = tid & 31, half = lane >> 4;
  int mBlk = blockIdx.x * 128;
  int ntg = blockIdx.y * 8 + wave;

  v8f acc[8];
  v8f z = {0.f, 0.f, 0.f, 0.f, 0.f, 0.f, 0.f, 0.f};
#pragma unroll
  for (int i = 0; i < 8; ++i) acc[i] = z;

  const __bf16* tbase =
      Tp + (size_t)blockIdx.x * 8 * 32 * 32 * 16 + (size_t)lane * 16;

  for (int kb = 0; kb < 32; ++kb) {
    v16bf bfr = *(const v16bf*)(Wp + ((size_t)(ntg * 32 + kb) * 32 + lane) * 16);
    v16bf a_cur = *(const v16bf*)(tbase + ((size_t)0 * 32 + kb) * 32 * 16);
    v16bf a_nxt;
#pragma unroll
    for (int mt = 0; mt < 8; ++mt) {
      if (mt < 7)
        a_nxt = *(const v16bf*)(tbase + ((size_t)(mt + 1) * 32 + kb) * 32 * 16);
      acc[mt] = __builtin_amdgcn_wmma_f32_16x16x32_bf16(
          false, a_cur, false, bfr, (short)0, acc[mt], false, false);
      a_cur = a_nxt;
    }
  }

  int ncol = ntg * 16 + (lane & 15);
  float bv = bias[ncol];
  float ev = dirs[ncol], cv = dirs[DIMX + ncol], nv = dirs[2 * DIMX + ncol];
#pragma unroll
  for (int mt = 0; mt < 8; ++mt) {
    int mbase = mBlk + mt * 16 + half * 8;
#pragma unroll
    for (int r = 0; r < 8; ++r) {
      int m = mbase + r;
      float ge = Ge[m], gc = Gc[m], gn = Gn[m];
      float alpha = 1.f - (ge + gc + gn);
      float ho = Hin[(size_t)m * DIMX + ncol];
      float u = ho * alpha + acc[mt][r] + bv + ge * ev + gc * cv + gn * nv;
      Hout[(size_t)m * DIMX + ncol] = u;
    }
  }
}

// ---------------------------------------------------------------------------
extern "C" void kernel_launch(void* const* d_in, const int* in_sizes, int n_in,
                              void* d_out, int out_size, void* d_ws,
                              size_t ws_size, hipStream_t stream) {
  const float* h0 = (const float*)d_in[0];
  const float* W1 = (const float*)d_in[1];
  const float* b1 = (const float*)d_in[2];
  const float* W2 = (const float*)d_in[3];
  const float* b2 = (const float*)d_in[4];
  const float* ae = (const float*)d_in[5];
  const float* ac = (const float*)d_in[6];
  const float* an = (const float*)d_in[7];
  float* out = (float*)d_out;

  char* ws = (char*)d_ws;
  size_t off = 0;
  float*  Hbuf = (float*)(ws + off);  off += (size_t)BATCH * DIMX * 4;  // 256 MB
  __bf16* Tp   = (__bf16*)(ws + off); off += (size_t)BATCH * DIMX * 2;  // 128 MB
  __bf16* W1p  = (__bf16*)(ws + off); off += (size_t)DIMX * DIMX * 2;
  __bf16* W2p  = (__bf16*)(ws + off); off += (size_t)DIMX * DIMX * 2;
  float*  dirs = (float*)(ws + off);  off += 3 * DIMX * 4;
  float*  Ge   = (float*)(ws + off);  off += (size_t)BATCH * 4;
  float*  Gc   = (float*)(ws + off);  off += (size_t)BATCH * 4;
  float*  Gn   = (float*)(ws + off);  off += (size_t)BATCH * 4;

  prep_dirs<<<3, 256, 0, stream>>>(ae, ac, an, dirs);
  pack_w<<<dim3(256, 2), 256, 0, stream>>>(W1, W2, W1p, W2p);

  // stats + coefficients for layer 0 (reads h0, no clamp, no writes to h0)
  rowpass<<<BATCH / 8, 256, 0, stream>>>((float*)h0, dirs, Ge, Gc, Gn, out,
                                         /*traceLayer=*/0, /*doClamp=*/0);

  float* hl = (float*)h0;
  for (int l = 0; l < NLAYERS; ++l) {
    // h1,h3,h5 -> ws buffer; h2,h4,h6 -> d_out h_final region (h6 last)
    float* hnext = ((l + 1) & 1) ? Hbuf : out;
    gemm1_tanh<<<dim3(BATCH / 128, DIMX / 128), 256, 0, stream>>>(hl, W1p, b1, Tp);
    gemm2_epi<<<dim3(BATCH / 128, DIMX / 128), 256, 0, stream>>>(
        Tp, W2p, b2, hl, dirs, Ge, Gc, Gn, hnext);
    int tl = (l + 1 <= NLAYERS - 1) ? (l + 1) : -1;
    rowpass<<<BATCH / 8, 256, 0, stream>>>(hnext, dirs, Ge, Gc, Gn, out, tl,
                                           /*doClamp=*/1);
    hl = hnext;
  }
}
